// CustomConv2d_21036749816009
// MI455X (gfx1250) — matrix operand order repaired and checked by reading, not compiled, not run on hardware
//
#include <hip/hip_runtime.h>
#include <hip/hip_bf16.h>

// ---------------------------------------------------------------------------
// Problem constants (from the reference): B=2, N=50000, K=16, CIN=64, COUT=32,
// M=4.  NT = B*N flattened nodes.
// ---------------------------------------------------------------------------
#define NB     2
#define NN     50000
#define NT     100000      // B*N
#define KNBR   16
#define CIN    64
#define COUT   32
#define MMIX   4
#define NCOLS  144         // 128 (W, m-major) + 4 (u) + 4 (v) + 8 pad
#define XS     72          // LDS row stride (bf16 elems) for x tile, padded
#define BS     72          // LDS row stride (bf16 elems) for Wcat, padded

typedef __bf16 bf16;
typedef __attribute__((ext_vector_type(16))) __bf16 v16bf;
typedef __attribute__((ext_vector_type(8)))  __bf16 bf16x8;
typedef __attribute__((ext_vector_type(8)))  float  v8f;

// ---------------------------------------------------------------------------
// Kernel 1: WMMA GEMM.  wxperm[node][o][m] = sum_c x[node][c] * W[m][o][c]
//           plus ux[node][m] = u_m . x_node and vx[node][m] = v_m . x_node,
//           produced as extra columns of a single [64 x 144] B matrix.
// Block = 256 threads = 8 waves; each wave owns a 16-node row tile.
// ---------------------------------------------------------------------------
__global__ __launch_bounds__(256)
void k1_wmma_transform(const float* __restrict__ x,
                       const float* __restrict__ weight,
                       const float* __restrict__ u,
                       const float* __restrict__ v,
                       float* __restrict__ wxperm,
                       float* __restrict__ uxa,
                       float* __restrict__ vxa) {
    __shared__ __attribute__((aligned(16))) bf16 Xs[128 * XS];
    __shared__ __attribute__((aligned(16))) bf16 Bs[NCOLS * BS];

    const int  tid       = threadIdx.x;
    const int  base      = blockIdx.x * 128;
    const bool fullBlock = (base + 128) <= NT;   // uniform: 781 of 782 blocks

    // Stage Wcat (144 cols x 64 rows, column-major [col][c]) as bf16.
    for (int i = tid; i < NCOLS * CIN; i += 256) {
        int cc = i >> 6, c = i & 63;
        float val;
        if (cc < 128)      val = weight[cc * CIN + c];        // col = m*32+o
        else if (cc < 132) val = u[(cc - 128) * CIN + c];
        else if (cc < 136) val = v[(cc - 132) * CIN + c];
        else               val = 0.0f;
        Bs[cc * BS + c] = (bf16)val;
    }
    // Stage 128 node rows of x as bf16 (guard only in the single tail block).
    if (fullBlock) {
        for (int i = tid; i < 128 * CIN; i += 256) {
            int r = i >> 6, c = i & 63;
            Xs[r * XS + c] = (bf16)x[(base + r) * CIN + c];
        }
    } else {
        for (int i = tid; i < 128 * CIN; i += 256) {
            int r = i >> 6, c = i & 63;
            int node = base + r;
            float val = (node < NT) ? x[node * CIN + c] : 0.0f;
            Xs[r * XS + c] = (bf16)val;
        }
    }
    __syncthreads();

    const int lane = tid & 31;
    const int wid  = tid >> 5;
    const int rowA = (wid << 4) + (lane & 15);
    const int kb   = (lane < 16) ? 0 : 8;      // 16-bit A-frag K base per half

    // Build the two A fragments (k-step 0: K=0..31, k-step 1: K=32..63).
    const bf16* xr = &Xs[rowA * XS];
    v16bf a0, a1;
    {
        bf16x8 lo = *(const bf16x8*)(xr + kb);
        bf16x8 hi = *(const bf16x8*)(xr + kb + 16);
        for (int i = 0; i < 8; ++i) { a0[i] = lo[i]; a0[8 + i] = hi[i]; }
        lo = *(const bf16x8*)(xr + kb + 32);
        hi = *(const bf16x8*)(xr + kb + 48);
        for (int i = 0; i < 8; ++i) { a1[i] = lo[i]; a1[8 + i] = hi[i]; }
    }

    const int colB    = lane & 15;
    const int kbB     = (lane < 16) ? 0 : 8;   // 16-bit B-frag K base per half
    const int nodeRow = base + (wid << 4) + ((lane < 16) ? 0 : 8);

    for (int t = 0; t < 9; ++t) {
        const bf16* bc = &Bs[(t * 16 + colB) * BS];
        v16bf b0, b1;
        {
            bf16x8 p0 = *(const bf16x8*)(bc + kbB);
            bf16x8 p1 = *(const bf16x8*)(bc + kbB + 16);
            for (int i = 0; i < 8; ++i) { b0[i] = p0[i]; b0[8 + i] = p1[i]; }
            p0 = *(const bf16x8*)(bc + 32 + kbB);
            p1 = *(const bf16x8*)(bc + 32 + kbB + 16);
            for (int i = 0; i < 8; ++i) { b1[i] = p0[i]; b1[8 + i] = p1[i]; }
        }

        v8f acc = {0.f, 0.f, 0.f, 0.f, 0.f, 0.f, 0.f, 0.f};
        acc = __builtin_amdgcn_wmma_f32_16x16x32_bf16(
                  false, a0, false, b0, (short)0, acc, false, false);
        acc = __builtin_amdgcn_wmma_f32_16x16x32_bf16(
                  false, a1, false, b1, (short)0, acc, false, false);

        // Scatter C tile: lane holds col cc = t*16 + (lane&15); VGPR g holds
        // row M = g (lanes 0-15) or 8+g (lanes 16-31).
        const int cc = t * 16 + colB;
        if (fullBlock) {
            // Common path: no per-element range guard.
            for (int g = 0; g < 8; ++g) {
                const int   node = nodeRow + g;
                const float valf = acc[g];
                if (cc < 128)      wxperm[node * 128 + (cc & 31) * MMIX + (cc >> 5)] = valf;
                else if (cc < 132) uxa[node * MMIX + (cc - 128)] = valf;
                else if (cc < 136) vxa[node * MMIX + (cc - 132)] = valf;
            }
        } else {
            for (int g = 0; g < 8; ++g) {
                const int node = nodeRow + g;
                if (node < NT) {
                    const float valf = acc[g];
                    if (cc < 128)      wxperm[node * 128 + (cc & 31) * MMIX + (cc >> 5)] = valf;
                    else if (cc < 132) uxa[node * MMIX + (cc - 128)] = valf;
                    else if (cc < 136) vxa[node * MMIX + (cc - 132)] = valf;
                }
            }
        }
    }
}

// ---------------------------------------------------------------------------
// Kernel 2: softmax-gated neighbor gather-accumulate.
// Phase A: q[node][k][m] = softmax_m(vx[adj] + ux + c) into LDS (+ adj cache).
// Phase B: lane = output channel o; one coalesced float4 load of
//          wxperm[neighbor][o][0..3] per (node,k), 4 FMAs, then *1/deg + bias.
// ---------------------------------------------------------------------------
__global__ __launch_bounds__(256)
void k2_gather(const int*   __restrict__ adj,
               const float* __restrict__ bias,
               const float* __restrict__ cvec,
               const float* __restrict__ wxperm,
               const float* __restrict__ uxa,
               const float* __restrict__ vxa,
               float*       __restrict__ out) {
    __shared__ float4 qs[128][KNBR];
    __shared__ int    js[128][KNBR];

    const int  tid       = threadIdx.x;
    const int  lane      = tid & 31;
    const int  wid       = tid >> 5;
    const int  base      = blockIdx.x * 128;
    const bool fullBlock = (base + 128) <= NT;

    const float c0 = cvec[0], c1 = cvec[1], c2 = cvec[2], c3 = cvec[3];

    // ---- Phase A: gates.  Lanes cover (2 nodes) x (16 neighbor slots). ----
    const int sub = lane >> 4;     // which node of the pair
    const int k   = lane & 15;     // neighbor slot
    for (int it = 0; it < 8; ++it) {
        const int nl   = (wid << 4) + it * 2 + sub;
        const int node = base + nl;
        int   j  = 0;
        float t0 = c0, t1 = c1, t2 = c2, t3 = c3;
        if (fullBlock || node < NT) {
            j = adj[node * KNBR + k];
            const float4 ug = *(const float4*)&uxa[node * MMIX];
            t0 += ug.x; t1 += ug.y; t2 += ug.z; t3 += ug.w;
            if (j != 0) {
                const int row = (node / NN) * NN + j - 1;  // batch-local index
                const float4 vg = *(const float4*)&vxa[row * MMIX];
                t0 += vg.x; t1 += vg.y; t2 += vg.z; t3 += vg.w;
            }
        }
        const float mx = fmaxf(fmaxf(t0, t1), fmaxf(t2, t3));
        const float e0 = __expf(t0 - mx), e1 = __expf(t1 - mx);
        const float e2 = __expf(t2 - mx), e3 = __expf(t3 - mx);
        const float inv = 1.0f / (e0 + e1 + e2 + e3);
        qs[nl][k] = make_float4(e0 * inv, e1 * inv, e2 * inv, e3 * inv);
        js[nl][k] = j;
    }
    __syncthreads();

    // ---- Phase B: weighted gather.  lane == output channel o. ----
    const float bl = bias[lane];
    for (int p = 0; p < 16; ++p) {
        const int nl   = (wid << 4) + p;
        const int node = base + nl;
        if (node >= NT) break;                    // uniform across the wave
        const int rowBase = (node / NN) * NN * 128 - 128;  // (b*N + j-1)*128
        float acc = 0.0f;
        int   deg = 0;
        for (int kk = 0; kk < KNBR; ++kk) {
            const int j = js[nl][kk];
            if (j == 0) continue;                 // empty slot contributes 0
            ++deg;
            const float4 q  = qs[nl][kk];
            const float4 wv = *(const float4*)&wxperm[rowBase + j * 128 + lane * 4];
            acc = fmaf(q.x, wv.x, acc);
            acc = fmaf(q.y, wv.y, acc);
            acc = fmaf(q.z, wv.z, acc);
            acc = fmaf(q.w, wv.w, acc);
        }
        const float invd = deg ? (1.0f / (float)deg) : 0.0f;
        out[node * COUT + lane] = acc * invd + bl;
    }
}

// ---------------------------------------------------------------------------
// Launch.  d_in order: x, adj, weight, bias, u, v, c.
// Workspace: wxperm (100000*128 f32) | ux (100000*4) | vx (100000*4) ~ 54.4 MB
// ---------------------------------------------------------------------------
extern "C" void kernel_launch(void* const* d_in, const int* in_sizes, int n_in,
                              void* d_out, int out_size, void* d_ws, size_t ws_size,
                              hipStream_t stream) {
    const float* x      = (const float*)d_in[0];
    const int*   adj    = (const int*)  d_in[1];
    const float* weight = (const float*)d_in[2];
    const float* bias   = (const float*)d_in[3];
    const float* u      = (const float*)d_in[4];
    const float* v      = (const float*)d_in[5];
    const float* c      = (const float*)d_in[6];
    float*       out    = (float*)d_out;

    float* ws     = (float*)d_ws;
    float* wxperm = ws;                          // 12,800,000 f32
    float* uxa    = ws + (size_t)NT * 128;       //    400,000 f32
    float* vxa    = uxa + (size_t)NT * MMIX;     //    400,000 f32

    const int blocks = (NT + 127) / 128;         // 782

    k1_wmma_transform<<<blocks, 256, 0, stream>>>(x, weight, u, v,
                                                  wxperm, uxa, vxa);
    k2_gather<<<blocks, 256, 0, stream>>>(adj, bias, c,
                                          wxperm, uxa, vxa, out);
}
